// AutoCorrelation_90469191123617
// MI455X (gfx1250) — matrix operand order repaired and checked by reading, not compiled, not run on hardware
//
#include <hip/hip_runtime.h>
#include <hip/hip_bf16.h>
#include <math.h>

// ---------------------------------------------------------------------------
// AutoCorrelation attention for MI455X (gfx1250, wave32, WMMA)
// B=8, L=4096, D=1024, H=16, DK=64, top_k=8
// ---------------------------------------------------------------------------

#define Bb   8
#define Ll   4096
#define Dd   1024
#define Hh   16
#define DKk  64
#define TOPK 8

typedef __attribute__((ext_vector_type(16))) __bf16 v16bf;
typedef __attribute__((ext_vector_type(8)))  __bf16 v8bf;
typedef __attribute__((ext_vector_type(8)))  float  v8f;

// ============================================================================
// GEMM: out = X @ W^T + bias     X:(M,K) f32, W:(N,K) f32, bf16 WMMA, f32 acc
// transposed==0: out[row*N + col]          (row = b*L+t)
// transposed==1: out[(b*N + col)*L + t]    (channel-major for FFT stage)
// Block: 256 threads = 8 waves (2x4). Block tile 128x128, wave tile 64x32.
// Double-buffered LDS; next tile's global loads issued under current WMMAs.
// ============================================================================
#define TM 128
#define TN 128
#define TK 32

__global__ __launch_bounds__(256)
void gemm_bf16_wmma(const float* __restrict__ X, const float* __restrict__ W,
                    const float* __restrict__ bias, float* __restrict__ out,
                    int M, int Kd, int N, int transposed)
{
    __shared__ __bf16 Als[2][TM][TK + 8];
    __shared__ __bf16 Bls[2][TN][TK + 8];

    const int tid  = threadIdx.x;
    const int wave = tid >> 5;
    const int lane = tid & 31;
    const int m0   = blockIdx.y * TM;
    const int n0   = blockIdx.x * TN;

    const int wm = (wave >> 2) * 64;   // 0 or 64
    const int wn = (wave & 3) * 32;    // 0,32,64,96
    const int r  = lane & 15;

    // per-thread staging slots: 4 float4 per operand per tile
    int srow[4], sc4[4];
#pragma unroll
    for (int s = 0; s < 4; ++s) {
        const int slot = tid + s * 256;    // 1024 float4 slots
        srow[s] = slot >> 3;               // 8 float4 per 32-wide row
        sc4[s]  = (slot & 7) * 4;
    }

    v8f acc[4][2];
#pragma unroll
    for (int mt = 0; mt < 4; ++mt)
#pragma unroll
        for (int nt = 0; nt < 2; ++nt)
            acc[mt][nt] = (v8f){0.f,0.f,0.f,0.f,0.f,0.f,0.f,0.f};

    // ---- preload tile k0=0 into registers (loads issued back-to-back) ----
    float4 ra[4], rb[4];
#pragma unroll
    for (int s = 0; s < 4; ++s) {
        ra[s] = *(const float4*)(X + (size_t)(m0 + srow[s]) * Kd + sc4[s]);
        rb[s] = *(const float4*)(W + (size_t)(n0 + srow[s]) * Kd + sc4[s]);
    }

    int buf = 0;
    for (int k0 = 0; k0 < Kd; k0 += TK) {
        // ---- convert staged registers -> LDS[buf] as bf16 ----
#pragma unroll
        for (int s = 0; s < 4; ++s) {
            __bf16* da = &Als[buf][srow[s]][sc4[s]];
            da[0] = (__bf16)ra[s].x; da[1] = (__bf16)ra[s].y;
            da[2] = (__bf16)ra[s].z; da[3] = (__bf16)ra[s].w;
            __bf16* db = &Bls[buf][srow[s]][sc4[s]];
            db[0] = (__bf16)rb[s].x; db[1] = (__bf16)rb[s].y;
            db[2] = (__bf16)rb[s].z; db[3] = (__bf16)rb[s].w;
        }
        __syncthreads();

        // ---- issue next tile's global loads (retire under the WMMAs) ----
        if (k0 + TK < Kd) {
#pragma unroll
            for (int s = 0; s < 4; ++s) {
                ra[s] = *(const float4*)(X + (size_t)(m0 + srow[s]) * Kd + k0 + TK + sc4[s]);
                rb[s] = *(const float4*)(W + (size_t)(n0 + srow[s]) * Kd + k0 + TK + sc4[s]);
            }
        }

        // ---- fragments (documented 16-bit A/B VGPR layouts) + 8 WMMAs ----
        const int kbA = (lane < 16) ? 0 : 8;    // A: lanes0-15 K0..7,16..23
        const int kbB = (lane < 16) ? 0 : 8;    // B: lanes0-15 K0..7,16..23
        v16bf afrag[4], bfrag[2];
#pragma unroll
        for (int mt = 0; mt < 4; ++mt) {
            const __bf16* p = &Als[buf][wm + mt * 16 + r][0];
            v8bf lo = *(const v8bf*)(p + kbA);
            v8bf hi = *(const v8bf*)(p + 16 + kbA);
            afrag[mt] = __builtin_shufflevector(lo, hi,
                0,1,2,3,4,5,6,7,8,9,10,11,12,13,14,15);
        }
#pragma unroll
        for (int nt = 0; nt < 2; ++nt) {
            const __bf16* p = &Bls[buf][wn + nt * 16 + r][0];
            v8bf lo = *(const v8bf*)(p + kbB);
            v8bf hi = *(const v8bf*)(p + 16 + kbB);
            bfrag[nt] = __builtin_shufflevector(lo, hi,
                0,1,2,3,4,5,6,7,8,9,10,11,12,13,14,15);
        }

#pragma unroll
        for (int mt = 0; mt < 4; ++mt)
#pragma unroll
            for (int nt = 0; nt < 2; ++nt)
                acc[mt][nt] = __builtin_amdgcn_wmma_f32_16x16x32_bf16(
                    false, afrag[mt], false, bfrag[nt],
                    (short)0, acc[mt][nt], false, false);

        buf ^= 1;
    }

    // ---- epilogue: bias add + store ----
#pragma unroll
    for (int nt = 0; nt < 2; ++nt) {
        const int col = n0 + wn + nt * 16 + r;
        const float bv = bias[col];
#pragma unroll
        for (int mt = 0; mt < 4; ++mt) {
            v8f a = acc[mt][nt];
            const int rbase = m0 + wm + mt * 16 + ((lane < 16) ? 0 : 8);
            if (!transposed) {
#pragma unroll
                for (int v = 0; v < 8; ++v)
                    out[(size_t)(rbase + v) * N + col] = a[v] + bv;
            } else {
                const int bq = rbase / Ll;
                const int t  = rbase & (Ll - 1);
                float* o = out + ((size_t)bq * N + col) * (size_t)Ll + t;
                float4 lo = make_float4(a[0]+bv, a[1]+bv, a[2]+bv, a[3]+bv);
                float4 hi = make_float4(a[4]+bv, a[5]+bv, a[6]+bv, a[7]+bv);
                *(float4*)(o)     = lo;
                *(float4*)(o + 4) = hi;
            }
        }
    }
}

// ============================================================================
// FFT correlation: corr[bh][tau] = (1/(L*DK)) * real(ifft( sum_dk Qf.conj(Kf) ))
// One workgroup per (b,h). Forward DIF (natural->bitrev), product in bitrev
// order, inverse DIT (bitrev->natural): no bit-reversal permutation needed.
// Channel loads use the gfx1250 async global->LDS engine (ASYNCcnt): real
// parts DMA straight into the .x halves of the LDS complex buffers while the
// wave zeroes the .y halves with plain ds_stores (disjoint addresses).
// ============================================================================
__device__ __forceinline__ float2 cadd(float2 a, float2 b){ return make_float2(a.x+b.x, a.y+b.y); }
__device__ __forceinline__ float2 csub(float2 a, float2 b){ return make_float2(a.x-b.x, a.y-b.y); }
__device__ __forceinline__ float2 cmul(float2 a, float2 b){ return make_float2(a.x*b.x - a.y*b.y, a.x*b.y + a.y*b.x); }

#define FFT_THREADS 512

__global__ __launch_bounds__(FFT_THREADS)
void fft_corr(const float* __restrict__ QT, const float* __restrict__ KT,
              float* __restrict__ corr)
{
    __shared__ float2 SA[Ll];        // 32 KB: Q working buffer
    __shared__ float2 SB[Ll];        // 32 KB: K working buffer
    __shared__ float2 SP[Ll];        // 32 KB: spectrum accumulator
    __shared__ float2 TW[Ll / 2];    // 16 KB: twiddles exp(-2*pi*i*j/L)

    const int bh  = blockIdx.x;      // b*H + h
    const int tid = threadIdx.x;

    for (int j = tid; j < Ll / 2; j += FFT_THREADS) {
        float s, c;
        __sincosf((float)j * (6.283185307179586f / (float)Ll), &s, &c);
        TW[j] = make_float2(c, -s);
    }
    for (int j = tid; j < Ll; j += FFT_THREADS) SP[j] = make_float2(0.f, 0.f);
    __syncthreads();

    const unsigned saBase = (unsigned)(uintptr_t)&SA[0];
    const unsigned sbBase = (unsigned)(uintptr_t)&SB[0];

    for (int dk = 0; dk < DKk; ++dk) {
        const size_t chan = ((size_t)bh * DKk + dk) * (size_t)Ll;

        // ---- async DMA real parts into LDS (.x), zero imag (.y) in LDS ----
#pragma unroll
        for (int s = 0; s < Ll / FFT_THREADS; ++s) {
            const int j = tid + s * FFT_THREADS;
            const unsigned da = saBase + (unsigned)j * 8u;
            const unsigned db = sbBase + (unsigned)j * 8u;
            const unsigned long long ga = (unsigned long long)(QT + chan + j);
            const unsigned long long gb = (unsigned long long)(KT + chan + j);
            asm volatile("global_load_async_to_lds_b32 %0, %1, off"
                         :: "v"(da), "v"(ga) : "memory");
            asm volatile("global_load_async_to_lds_b32 %0, %1, off"
                         :: "v"(db), "v"(gb) : "memory");
            SA[j].y = 0.f;
            SB[j].y = 0.f;
        }
        asm volatile("s_wait_asynccnt 0" ::: "memory");
        __syncthreads();

        // ---- fused forward DIF FFTs of Q and K ----
        for (int h = Ll / 2; h >= 1; h >>= 1) {
            const int step = (Ll / 2) / h;
            for (int idx = tid; idx < Ll / 2; idx += FFT_THREADS) {
                const int i = idx & (h - 1);
                const int u = ((idx & ~(h - 1)) << 1) + i;
                const int v = u + h;
                const float2 w = TW[i * step];
                float2 qa = SA[u], qb = SA[v];
                SA[u] = cadd(qa, qb);
                SA[v] = cmul(csub(qa, qb), w);
                float2 ka = SB[u], kb = SB[v];
                SB[u] = cadd(ka, kb);
                SB[v] = cmul(csub(ka, kb), w);
            }
            __syncthreads();
        }

        // ---- accumulate P += Qf * conj(Kf)  (bit-reversed order, consistent)
        for (int j = tid; j < Ll; j += FFT_THREADS) {
            const float2 qf = SA[j], kf = SB[j];
            float2 p = SP[j];
            p.x += qf.x * kf.x + qf.y * kf.y;
            p.y += qf.y * kf.x - qf.x * kf.y;
            SP[j] = p;
        }
        __syncthreads();
    }

    // ---- inverse DIT on bit-reversed spectrum -> natural-order lags ----
    for (int h = 1; h <= Ll / 2; h <<= 1) {
        const int step = (Ll / 2) / h;
        for (int idx = tid; idx < Ll / 2; idx += FFT_THREADS) {
            const int i = idx & (h - 1);
            const int u = ((idx & ~(h - 1)) << 1) + i;
            const int v = u + h;
            const float2 w  = TW[i * step];      // use conj(w): exp(+i*ang)
            const float2 pv = SP[v];
            const float2 t  = make_float2(pv.x * w.x + pv.y * w.y,
                                          pv.y * w.x - pv.x * w.y);
            const float2 a  = SP[u];
            SP[u] = cadd(a, t);
            SP[v] = csub(a, t);
        }
        __syncthreads();
    }

    const float scale = 1.0f / ((float)Ll * (float)DKk);
    for (int j = tid; j < Ll; j += FFT_THREADS)
        corr[(size_t)bh * Ll + j] = SP[j].x * scale;
}

// ============================================================================
// Top-8 + softmax per (b,h)
// ============================================================================
__global__ __launch_bounds__(256)
void topk_softmax(const float* __restrict__ corr,
                  float* __restrict__ topw, int* __restrict__ topi)
{
    __shared__ float vals[Ll];
    __shared__ float rmax[256];
    __shared__ int   ridx[256];
    __shared__ float swv[TOPK];
    __shared__ int   swi[TOPK];

    const int bh = blockIdx.x, tid = threadIdx.x;
    for (int j = tid; j < Ll; j += 256) vals[j] = corr[(size_t)bh * Ll + j];
    __syncthreads();

    for (int it = 0; it < TOPK; ++it) {
        float best = -3.0e38f; int bi = 0;
        for (int j = tid; j < Ll; j += 256)
            if (vals[j] > best) { best = vals[j]; bi = j; }
        rmax[tid] = best; ridx[tid] = bi;
        __syncthreads();
        for (int s = 128; s > 0; s >>= 1) {
            if (tid < s && rmax[tid + s] > rmax[tid]) {
                rmax[tid] = rmax[tid + s]; ridx[tid] = ridx[tid + s];
            }
            __syncthreads();
        }
        if (tid == 0) { swv[it] = rmax[0]; swi[it] = ridx[0]; vals[ridx[0]] = -3.0e38f; }
        __syncthreads();
    }
    if (tid == 0) {
        float mx = swv[0];
        for (int i = 1; i < TOPK; ++i) mx = fmaxf(mx, swv[i]);
        float e[TOPK], sum = 0.f;
        for (int i = 0; i < TOPK; ++i) { e[i] = __expf(swv[i] - mx); sum += e[i]; }
        for (int i = 0; i < TOPK; ++i) {
            topw[bh * TOPK + i] = e[i] / sum;
            topi[bh * TOPK + i] = swi[i];
        }
    }
}

// ============================================================================
// Time-delay aggregation: ctx[b,t,d] = sum_i w[h,i] * V[b,(t-idx[h,i])%L,d]
// One block per (b,t); threads cover d in float4 chunks (coalesced gathers).
// ============================================================================
__global__ __launch_bounds__(256)
void aggregate(const float* __restrict__ V, const float* __restrict__ topw,
               const int* __restrict__ topi, float* __restrict__ ctx)
{
    const int blk = blockIdx.x;           // b*L + t
    const int b = blk >> 12;
    const int t = blk & (Ll - 1);
    const int tid = threadIdx.x;

    __shared__ float w[Hh][TOPK];
    __shared__ int   ix[Hh][TOPK];
    if (tid < Hh * TOPK) {
        const int h = tid >> 3, i = tid & 7;
        w[h][i]  = topw[(b * Hh + h) * TOPK + i];
        ix[h][i] = topi[(b * Hh + h) * TOPK + i];
    }
    __syncthreads();

    const int d0 = tid * 4;
    const int h  = d0 >> 6;
    float4 s = make_float4(0.f, 0.f, 0.f, 0.f);
#pragma unroll
    for (int i = 0; i < TOPK; ++i) {
        const int ts = (t - ix[h][i]) & (Ll - 1);
        const float4 vv = *(const float4*)(V + ((size_t)b * Ll + ts) * Dd + d0);
        const float wi = w[h][i];
        s.x += wi * vv.x; s.y += wi * vv.y; s.z += wi * vv.z; s.w += wi * vv.w;
    }
    *(float4*)(ctx + (size_t)blk * Dd + d0) = s;
}

// ============================================================================
// launch
// ============================================================================
extern "C" void kernel_launch(void* const* d_in, const int* in_sizes, int n_in,
                              void* d_out, int out_size, void* d_ws, size_t ws_size,
                              hipStream_t stream)
{
    const float* query = (const float*)d_in[0];
    const float* key   = (const float*)d_in[1];
    const float* value = (const float*)d_in[2];
    const float* Wq = (const float*)d_in[3];  const float* bq = (const float*)d_in[4];
    const float* Wk = (const float*)d_in[5];  const float* bk = (const float*)d_in[6];
    const float* Wv = (const float*)d_in[7];  const float* bv = (const float*)d_in[8];
    const float* Wo = (const float*)d_in[9];  const float* bo = (const float*)d_in[10];

    const size_t BDL = (size_t)Bb * Dd * Ll;      // 33.5M floats per tensor
    float* ws   = (float*)d_ws;
    float* QT   = ws;                              // [b][h*64+dk][t]
    float* KT   = QT + BDL;
    float* Vn   = KT + BDL;                        // [b][t][d]
    float* CTX  = Vn + BDL;                        // [b][t][d]
    float* CORR = CTX + BDL;                       // [bh][tau]
    float* TOPW = CORR + (size_t)Bb * Hh * Ll;
    int*   TOPI = (int*)(TOPW + Bb * Hh * TOPK);

    const int M = Bb * Ll;                         // 32768
    dim3 ggrid(Dd / TN, M / TM);                   // (8, 256)

    gemm_bf16_wmma<<<ggrid, 256, 0, stream>>>(query, Wq, bq, QT, M, Dd, Dd, 1);
    gemm_bf16_wmma<<<ggrid, 256, 0, stream>>>(key,   Wk, bk, KT, M, Dd, Dd, 1);
    gemm_bf16_wmma<<<ggrid, 256, 0, stream>>>(value, Wv, bv, Vn, M, Dd, Dd, 0);

    fft_corr<<<Bb * Hh, FFT_THREADS, 0, stream>>>(QT, KT, CORR);
    topk_softmax<<<Bb * Hh, 256, 0, stream>>>(CORR, TOPW, TOPI);
    aggregate<<<Bb * Ll, 256, 0, stream>>>(Vn, TOPW, TOPI, CTX);

    gemm_bf16_wmma<<<ggrid, 256, 0, stream>>>(CTX, Wo, bo, (float*)d_out, M, Dd, Dd, 0);
}
